// MultiHeadDecoder_7739531067805
// MI455X (gfx1250) — matrix operand order repaired and checked by reading, not compiled, not run on hardware
//
#include <hip/hip_runtime.h>
#include <hip/hip_bf16.h>

typedef _Float16 v16h __attribute__((ext_vector_type(16)));
typedef _Float16 v8h  __attribute__((ext_vector_type(8)));
typedef float    v8f  __attribute__((ext_vector_type(8)));

#define Bn   16
#define Nn   500
#define Dn   128
#define Hn   8
#define KDn  16
#define KPAD 32          // K-dim padded to 32 in memory -> fragment loads need no fixup
#define NPAD 512
#define NTI  32          // 512/16 tiles per dim
#define NORMC 0.25f      // 1/sqrt(16)

__device__ __forceinline__ v8f zero8f() {
    v8f r; for (int i = 0; i < 8; ++i) r[i] = 0.f; return r;
}
__device__ __forceinline__ v16h cat16(v8h lo, v8h hi) {
    v16h r;
    for (int e = 0; e < 8; ++e) { r[e] = lo[e]; r[8 + e] = hi[e]; }
    return r;
}
// intra-wave LDS ordering (LDS is in-order per wave; this only stops compiler reordering)
__device__ __forceinline__ void wave_sync() {
    __builtin_amdgcn_wave_barrier();
    __builtin_amdgcn_fence(__ATOMIC_ACQ_REL, "wavefront");
    __builtin_amdgcn_wave_barrier();
}

// ---------------------------------------------------------------- stage 1
// g[br][b][d] = max_n x[b][n][d];  br 0 = pos_em, 1 = h_em (node)
__global__ void k_maxpool(const float* __restrict__ h_em,
                          const float* __restrict__ pos_em,
                          float* __restrict__ g) {
    int idx = blockIdx.x * 256 + threadIdx.x;          // 2*B*D = 4096
    if (idx >= 2 * Bn * Dn) return;
    int br = idx / (Bn * Dn);
    int r  = idx % (Bn * Dn);
    int b = r / Dn, d = r % Dn;
    const float* x = br ? h_em : pos_em;
    const float* xp = x + ((long)b * Nn) * Dn + d;
    float m = xp[0];
    for (int n = 1; n < Nn; ++n) { float v = xp[(long)n * Dn]; m = v > m ? v : m; }
    g[idx] = m;
}

// ---------------------------------------------------------------- stage 2
// gp[br][b][d] = sum_{d2} g[br][b][d2] * P_graph[br][d2][d]
__global__ void k_gp(const float* __restrict__ g,
                     const float* __restrict__ Pg_pos,
                     const float* __restrict__ Pg_node,
                     float* __restrict__ gp) {
    int idx = blockIdx.x * 256 + threadIdx.x;          // 4096
    if (idx >= 2 * Bn * Dn) return;
    int br = idx / (Bn * Dn);
    int r  = idx % (Bn * Dn);
    int b = r / Dn, d = r % Dn;
    const float* P  = br ? Pg_node : Pg_pos;
    const float* gv = g + (br * Bn + b) * Dn;
    float acc = 0.f;
    for (int d2 = 0; d2 < Dn; ++d2) acc += gv[d2] * P[d2 * Dn + d];
    gp[idx] = acc;
}

// ---------------------------------------------------------------- stage 3
// weff[m][h][d][k] = sum_{d2} P_node[br][d][d2] * W_m[h][d2][k]
// bias[m][b][h][k] = sum_{d}  gp[br][b][d]      * W_m[h][d][k]
// m: 0=Qpos 1=Kpos 2=Qnode 3=Knode ; br = m>>1
__global__ void k_weff_bias(const float* __restrict__ Pn_pos,
                            const float* __restrict__ Pn_node,
                            const float* __restrict__ Wq_pos,
                            const float* __restrict__ Wk_pos,
                            const float* __restrict__ Wq_node,
                            const float* __restrict__ Wk_node,
                            const float* __restrict__ gp,
                            float* __restrict__ weff,
                            float* __restrict__ bias) {
    const int NW = 4 * Hn * Dn * KDn;                  // 65536
    const int NB = 4 * Bn * Hn * KDn;                  // 8192
    int idx = blockIdx.x * 256 + threadIdx.x;
    if (idx < NW) {
        int m = idx / (Hn * Dn * KDn);
        int r = idx % (Hn * Dn * KDn);
        int h = r / (Dn * KDn); r %= Dn * KDn;
        int d = r / KDn, k = r % KDn;
        const float* P = (m < 2) ? Pn_pos : Pn_node;
        const float* W = (m == 0) ? Wq_pos : (m == 1) ? Wk_pos
                        : (m == 2) ? Wq_node : Wk_node;
        float acc = 0.f;
        for (int d2 = 0; d2 < Dn; ++d2)
            acc += P[d * Dn + d2] * W[(h * Dn + d2) * KDn + k];
        weff[idx] = acc;
    } else if (idx < NW + NB) {
        int i2 = idx - NW;
        int m = i2 / (Bn * Hn * KDn);
        int r = i2 % (Bn * Hn * KDn);
        int b = r / (Hn * KDn); r %= Hn * KDn;
        int h = r / KDn, k = r % KDn;
        const float* W = (m == 0) ? Wq_pos : (m == 1) ? Wk_pos
                        : (m == 2) ? Wq_node : Wk_node;
        const float* gpv = gp + ((m >> 1) * Bn + b) * Dn;
        float acc = 0.f;
        for (int d = 0; d < Dn; ++d)
            acc += gpv[d] * W[(h * Dn + d) * KDn + k];
        bias[i2] = acc;
    }
}

// ---------------------------------------------------------------- stage 4
// qk[m][b][h][n][kp] (f16), n padded to 512, k padded to 32, pads = 0
__global__ void k_qk(const float* __restrict__ h_em,
                     const float* __restrict__ pos_em,
                     const float* __restrict__ weff,
                     const float* __restrict__ bias,
                     _Float16* __restrict__ qk) {
    long idx = (long)blockIdx.x * 256 + threadIdx.x;   // 4*B*H*512*32 = 8388608
    if (idx >= (long)4 * Bn * Hn * NPAD * KPAD) return;
    int k = idx & (KPAD - 1);  long r = idx >> 5;
    int n = r & (NPAD - 1); r >>= 9;
    int h = r & 7;  r >>= 3;
    int b = r & 15; int m = (int)(r >> 4);
    if (k >= KDn || n >= Nn) { qk[idx] = (_Float16)0.f; return; }
    const float* x  = (m < 2) ? pos_em : h_em;
    const float* xr = x + ((long)b * Nn + n) * Dn;
    const float* wf = weff + ((m * Hn + h) * Dn) * KDn + k;
    float acc = bias[((m * Bn + b) * Hn + h) * KDn + k];
    for (int d = 0; d < Dn; ++d) acc += xr[d] * wf[d * KDn];
    qk[idx] = (_Float16)acc;
}

// ---------------------------------------------------------------- stage 5
// Fused compat (16 WMMA heads) + pairwise MLP (4 WMMA / row-tile) per 16x16 tile.
// One wave per tile; waves are fully independent (no block barriers).
__global__ __launch_bounds__(128)
void k_fused(const _Float16* __restrict__ qk,
             const float* __restrict__ W1, const float* __restrict__ b1,
             const float* __restrict__ W2, const float* __restrict__ b2,
             const float* __restrict__ W3, const float* __restrict__ b3p,
             float* __restrict__ out) {
    __shared__ __align__(16) _Float16 s_compat[4][16][16][16]; // [wave][m][n][head] 32KB
    __shared__ __align__(16) _Float16 s_h1[4][16][32];         // [wave][pair][feat] 4KB
    __shared__ __align__(16) float    s_h2[4][16][32];         // 8KB

    int tid  = threadIdx.x;
    int w    = tid >> 5;
    int lane = tid & 31;
    int ln   = lane & 15;
    bool hi  = lane >= 16;

    int wg  = blockIdx.x * 4 + w;          // 16384 wave-tasks
    int b   = wg >> 10;
    int rem = wg & 1023;
    int it  = rem >> 5, jt = rem & 31;
    int i0  = it * 16, j0 = jt * 16;

    // --- B fragments for W1 (K=16, rows 16..31 forced zero) and W2 (K=32) ---
    // B layout (32x16 f16): lane = column n, kbase = (lane<16 ? 0 : 8),
    // element e<8 = B[kbase+e][n], e>=8 = B[kbase+8+e][n].
    v16h w1f0, w1f1, w2f0, w2f1;
    int kb = hi ? 16 : 0;
    for (int e = 0; e < 16; ++e) {
        w1f0[e] = (e >= 8) ? (_Float16)0.f : (_Float16)W1[((kb >> 1) + e) * 32 + ln];
        w1f1[e] = (e >= 8) ? (_Float16)0.f : (_Float16)W1[((kb >> 1) + e) * 32 + 16 + ln];
        w2f0[e] = (_Float16)W2[((kb >> 1) + e + (e >= 8 ? 8 : 0)) * 32 + ln];
        w2f1[e] = (_Float16)W2[((kb >> 1) + e + (e >= 8 ? 8 : 0)) * 32 + 16 + ln];
    }
    float bias1a = b1[ln], bias1b = b1[16 + ln];
    float bias2a = b2[ln], bias2b = b2[16 + ln];
    float bias3  = b3p[0];

    // --- compat: one WMMA per head; K=16 real, memory pad supplies K=16..31=0 ---
    // A layout (16x32 f16): lane m=ln; e0..7 -> K=(hi?8:0)+e, e8..15 -> +16 (pad).
    // B layout (32x16 f16): lane n=ln; e0..7 -> K=(hi?8:0)+e, e8..15 -> +16 (pad).
    const int aoff = hi ? 8 : 0;
    const int boff = hi ? 8 : 0;
    const int woff = hi ? 8 : 0;    // D-fragment row offset

    auto load_frags = [&](int f, v16h& a, v16h& bf) {
        int qm = (f < 8) ? 0 : 2;
        int hh = f & 7;
        const _Float16* qp = qk
            + ((((long)qm * Bn + b) * Hn + hh) * NPAD + (i0 + ln)) * KPAD + aoff;
        const _Float16* kp = qk
            + ((((long)(qm + 1) * Bn + b) * Hn + hh) * NPAD + (j0 + ln)) * KPAD + boff;
        a  = cat16(*(const v8h*)qp, *(const v8h*)(qp + 16));
        bf = cat16(*(const v8h*)kp, *(const v8h*)(kp + 16));
    };

    v16h a_cur, b_cur;
    load_frags(0, a_cur, b_cur);
    for (int f = 0; f < 16; ++f) {
        v16h a_nxt, b_nxt;
        if (f < 15) load_frags(f + 1, a_nxt, b_nxt);
        v8f c = zero8f();
        c = __builtin_amdgcn_wmma_f32_16x16x32_f16(false, a_cur, false, b_cur,
                                                   (short)0, c, false, false);
        // D layout: lane col n=ln, VGPR v -> row v + woff
        for (int v = 0; v < 8; ++v)
            s_compat[w][v + woff][ln][f] = (_Float16)(c[v] * NORMC);
        if (f < 15) { a_cur = a_nxt; b_cur = b_nxt; }
    }
    wave_sync();

    // --- per row-tile MLP: rows = 16 pairs (i = i0+t fixed, j = j0+0..15) ---
    for (int t = 0; t < 16; ++t) {
        // layer 1: K = 16 heads; A elems 8..15 are don't-care (W1 rows 16..31 = 0)
        v8h ca = *(const v8h*)&s_compat[w][t][ln][aoff];
        v16h a1 = cat16(ca, ca);
        v8f h1a = zero8f(), h1b = zero8f();
        h1a = __builtin_amdgcn_wmma_f32_16x16x32_f16(false, a1, false, w1f0,
                                                     (short)0, h1a, false, false);
        h1b = __builtin_amdgcn_wmma_f32_16x16x32_f16(false, a1, false, w1f1,
                                                     (short)0, h1b, false, false);
        for (int v = 0; v < 8; ++v) {
            int row = v + woff;
            float x0 = h1a[v] + bias1a; x0 = x0 > 0.f ? x0 : 0.f;
            float x1 = h1b[v] + bias1b; x1 = x1 > 0.f ? x1 : 0.f;
            s_h1[w][row][ln]      = (_Float16)x0;
            s_h1[w][row][16 + ln] = (_Float16)x1;
        }
        wave_sync();

        // layer 2: K = 32 exact.  A elems 0..7 -> K=aoff+e, 8..15 -> K=aoff+16+e.
        v16h a2 = cat16(*(const v8h*)&s_h1[w][ln][aoff],
                        *(const v8h*)&s_h1[w][ln][aoff + 16]);
        v8f h2a = zero8f(), h2b = zero8f();
        h2a = __builtin_amdgcn_wmma_f32_16x16x32_f16(false, a2, false, w2f0,
                                                     (short)0, h2a, false, false);
        h2b = __builtin_amdgcn_wmma_f32_16x16x32_f16(false, a2, false, w2f1,
                                                     (short)0, h2b, false, false);
        for (int v = 0; v < 8; ++v) {
            int row = v + woff;
            float x0 = h2a[v] + bias2a; x0 = x0 > 0.f ? x0 : 0.f;
            float x1 = h2b[v] + bias2b; x1 = x1 > 0.f ? x1 : 0.f;
            s_h2[w][row][ln]      = x0;
            s_h2[w][row][16 + ln] = x1;
        }
        wave_sync();

        // layer 3: 32 -> 1 dot; lanes 0..15 own pair (i0+t, j0+ln).
        // W3[kk] is lane-uniform -> scalar (K$) loads.
        if (!hi) {
            int i = i0 + t, j = j0 + ln;
            if (i < Nn && j < Nn) {
                float acc = bias3;
                for (int kk = 0; kk < 32; ++kk)
                    acc += s_h2[w][ln][kk] * W3[kk];
                out[((long)b * Nn + i) * Nn + j] = acc;
            }
        }
        wave_sync();
    }
}

// ---------------------------------------------------------------- launch
extern "C" void kernel_launch(void* const* d_in, const int* in_sizes, int n_in,
                              void* d_out, int out_size, void* d_ws, size_t ws_size,
                              hipStream_t stream) {
    const float* h_em    = (const float*)d_in[0];
    const float* pos_em  = (const float*)d_in[1];
    // d_in[2] solving_state_for_net: unused by the reference
    const float* Wq_pos  = (const float*)d_in[3];
    const float* Wk_pos  = (const float*)d_in[4];
    const float* Wq_node = (const float*)d_in[5];
    const float* Wk_node = (const float*)d_in[6];
    const float* Pg_pos  = (const float*)d_in[7];
    const float* Pg_node = (const float*)d_in[8];
    const float* Pn_pos  = (const float*)d_in[9];
    const float* Pn_node = (const float*)d_in[10];
    const float* W1 = (const float*)d_in[11];
    const float* b1 = (const float*)d_in[12];
    const float* W2 = (const float*)d_in[13];
    const float* b2 = (const float*)d_in[14];
    const float* W3 = (const float*)d_in[15];
    const float* b3 = (const float*)d_in[16];
    float* out = (float*)d_out;

    // workspace (f32 counts): g 4096 | gp 4096 | weff 65536 | bias 8192 | qk f16 8388608
    float* g    = (float*)d_ws;
    float* gp   = g + 2 * Bn * Dn;
    float* weff = gp + 2 * Bn * Dn;
    float* bias = weff + 4 * Hn * Dn * KDn;
    _Float16* qk = (_Float16*)(bias + 4 * Bn * Hn * KDn);

    k_maxpool<<<(2 * Bn * Dn + 255) / 256, 256, 0, stream>>>(h_em, pos_em, g);
    k_gp<<<(2 * Bn * Dn + 255) / 256, 256, 0, stream>>>(g, Pg_pos, Pg_node, gp);
    k_weff_bias<<<(4 * Hn * Dn * KDn + 4 * Bn * Hn * KDn + 255) / 256, 256, 0, stream>>>(
        Pn_pos, Pn_node, Wq_pos, Wk_pos, Wq_node, Wk_node, gp, weff, bias);
    long nqk = (long)4 * Bn * Hn * NPAD * KPAD;
    k_qk<<<(int)((nqk + 255) / 256), 256, 0, stream>>>(h_em, pos_em, weff, bias, qk);
    int nblocks = (Bn * NTI * NTI) / 4;    // 4096 blocks, 4 waves each
    k_fused<<<nblocks, 128, 0, stream>>>(qk, W1, b1, W2, b2, W3, b3, out);
}
